// LEMCell_24532853195008
// MI455X (gfx1250) — matrix operand order repaired and checked by reading, not compiled, not run on hardware
//
#include <hip/hip_runtime.h>
#include <hip/hip_bf16.h>

// Fused LEM cell for MI455X (gfx1250, wave32, WMMA).
//
// One workgroup owns a 32-row batch tile: input GEMM once, then all 6
// recurrent updates entirely in LDS via v_wmma_f32_16x16x32_bf16 (bf16 in,
// f32 accum). HBM traffic is minimal: x,y,z read once, y,z written once
// (~320MB @ 23.3TB/s ~= 14us); the ~1MB bf16 transposed weights live in L2.
//
// Wave layout (8 waves / 256 threads): each wave owns a full N-slab and BOTH
// 16-row M-tiles. Per GEMM, the wave hoists ALL its A-fragments (32x256 tile,
// 16 x v16bf = 128 VGPRs) into registers ONCE, then streams only B (weight)
// fragments per N-tile: 1 B-load pair -> 2 WMMAs, and A-fragment LDS traffic
// is amortized across every N-tile (8x/6x/2x reduction vs per-tile reload).

#define NUM_UPDATES 6
#define MTILE 32              // batch rows per workgroup
#define UD 256                // hidden dim U == input dim D

typedef __bf16 bf16_t;
typedef __attribute__((ext_vector_type(16))) __bf16 v16bf;
typedef __attribute__((ext_vector_type(8)))  float  v8f;

union FragU { v16bf v; uint4 q[2]; };

// A-fragment (16x32 bf16, M x K) from row-major LDS, contiguous k order (same as B):
// lane<16 : row M=lane,    K = k0+0..15  (v0..v7)
// lane>=16: row M=lane-16, K = k0+16..31 (v0..v7)
__device__ __forceinline__ v16bf load_a_lds(const bf16_t* base, int row0,
                                            int stride, int k0, int lane) {
    const int half = lane >> 4;
    const int r    = lane & 15;
    const bf16_t* p = base + (row0 + r) * stride + k0 + half * 16;
    FragU f;
    f.q[0] = *(const uint4*)(p);        // 8 bf16 = 16B
    f.q[1] = *(const uint4*)(p + 8);    // 8 bf16 = 16B
    return f.v;
}

// B-fragment (32x16 bf16, K x N) from TRANSPOSED weights wT[N][256] row-major:
// lane<16 : col N=n0+lane,    K = k0+0..15  (2 per VGPR, consecutive)
// lane>=16: col N=n0+lane-16, K = k0+16..31
// => 32 contiguous bytes per lane: two b128 loads.
__device__ __forceinline__ v16bf load_b_wt(const bf16_t* wT, int n0, int k0,
                                           int lane) {
    const int half = lane >> 4;
    const int n    = lane & 15;
    const bf16_t* p = wT + (size_t)(n0 + n) * UD + k0 + half * 16;
    FragU f;
    f.q[0] = *(const uint4*)(p);
    f.q[1] = *(const uint4*)(p + 8);
    return f.v;
}

// C-fragment (16x16 f32): VGPR i, lane<16 -> D[m0+i][n0+lane];
// lane>=16 -> D[m0+8+i][n0+lane-16]
__device__ __forceinline__ void store_c_f32(float* dst, int m0, int n0,
                                            int stride, int lane, v8f c) {
    const int half = lane >> 4;
    const int n    = lane & 15;
#pragma unroll
    for (int i = 0; i < 8; ++i)
        dst[(m0 + i + 8 * half) * stride + n0 + n] = c[i];
}

__device__ __forceinline__ void store_c_bf16(bf16_t* dst, int m0, int n0,
                                             int stride, int lane, v8f c) {
    const int half = lane >> 4;
    const int n    = lane & 15;
#pragma unroll
    for (int i = 0; i < 8; ++i)
        dst[(m0 + i + 8 * half) * stride + n0 + n] = (bf16_t)c[i];
}

__device__ __forceinline__ float sigmoidf_(float a) {
    return 1.0f / (1.0f + __expf(-a));
}

// Tile GEMM: C[0:32, nbase:nbase+NT*16] = A[0:32, 0:256] @ W (W^T supplied).
// A-fragments register-resident across all NT N-tiles; B streamed from L2.
template <int NT, typename StoreF>
__device__ __forceinline__ void gemm_tile(const bf16_t* aLds, const bf16_t* wT,
                                          int nbase, int lane, StoreF&& store) {
    v16bf a0[8], a1[8];
#pragma unroll
    for (int k = 0; k < 8; ++k) {
        a0[k] = load_a_lds(aLds, 0, UD, k * 32, lane);
        a1[k] = load_a_lds(aLds, 16, UD, k * 32, lane);
    }
    for (int nt = 0; nt < NT; ++nt) {
        const int n0 = nbase + nt * 16;
        v8f acc0 = {}, acc1 = {};
#pragma unroll
        for (int k = 0; k < 8; ++k) {
            v16bf b = load_b_wt(wT, n0, k * 32, lane);
            acc0 = __builtin_amdgcn_wmma_f32_16x16x32_bf16(
                false, a0[k], false, b, (short)0, acc0, false, false);
            acc1 = __builtin_amdgcn_wmma_f32_16x16x32_bf16(
                false, a1[k], false, b, (short)0, acc1, false, false);
        }
        store(n0, 0, acc0);
        store(n0, 16, acc1);
    }
}

// ---------------------------------------------------------------------------
// Prologue: weights f32 -> bf16, transposed to [N][K=256] row-major so WMMA
// B-fragments are contiguous. ~1MB into d_ws, L2-resident.
// ws layout: wi[1024][256], wh[768][256], wt[256][256]  (bf16)
// ---------------------------------------------------------------------------
__global__ __launch_bounds__(256) void lem_prep_weights(
    const float* __restrict__ i2h,   // [256][1024]
    const float* __restrict__ h2h,   // [256][768]
    const float* __restrict__ tz,    // [256][256]
    bf16_t* __restrict__ wi, bf16_t* __restrict__ wh, bf16_t* __restrict__ wt) {
    int idx = blockIdx.x * 256 + threadIdx.x;   // 2048*256 threads total
    int n = idx >> 8;
    int k = idx & 255;
    if (n < 1024) {
        wi[idx] = (bf16_t)i2h[k * 1024 + n];
    } else if (n < 1024 + 768) {
        int nn = n - 1024;
        wh[nn * UD + k] = (bf16_t)h2h[k * 768 + nn];
    } else {
        int nn = n - (1024 + 768);
        wt[nn * UD + k] = (bf16_t)tz[k * UD + nn];
    }
}

// ---------------------------------------------------------------------------
// Main fused kernel. Dynamic LDS carve (288KB <= 320KB/WGP):
//   [  0..64 )KB  s_i    bf16 [32][1024]  input-GEMM result (i_dt1|i_dt2|i_z|i_y)
//   [ 64..96 )KB  s_y    f32  [32][256]
//   [ 96..128)KB  s_z    f32  [32][256]
//   [128..144)KB  s_ybf  bf16 [32][256]
//   [144..160)KB  s_zbf  bf16 [32][256]
//   [160..256)KB  s_th   f32  [32][768]   (aliased by s_xbf bf16 [32][256])
//   [256..288)KB  s_tz   f32  [32][256]
// ---------------------------------------------------------------------------
__global__ __launch_bounds__(256) void lem_main(
    const float* __restrict__ x, const float* __restrict__ dt,
    const float* __restrict__ y_in, const float* __restrict__ z_in,
    const bf16_t* __restrict__ wi, const bf16_t* __restrict__ wh,
    const bf16_t* __restrict__ wt,
    float* __restrict__ out_y, float* __restrict__ out_z) {

    extern __shared__ char smem[];
    bf16_t* s_i   = (bf16_t*)(smem);
    float*  s_y   = (float*)(smem + 64 * 1024);
    float*  s_z   = (float*)(smem + 96 * 1024);
    bf16_t* s_ybf = (bf16_t*)(smem + 128 * 1024);
    bf16_t* s_zbf = (bf16_t*)(smem + 144 * 1024);
    float*  s_th  = (float*)(smem + 160 * 1024);
    bf16_t* s_xbf = (bf16_t*)(smem + 160 * 1024);   // alias: dead after GEMM1
    float*  s_tz  = (float*)(smem + 256 * 1024);

    const int tid  = threadIdx.x;
    const int lane = tid & 31;
    const int wave = tid >> 5;              // 0..7: N-slab owner

    // pointwise mapping: each thread owns 32 contiguous elements of 32x256
    const int pr  = tid >> 3;               // row 0..31
    const int pc0 = (tid & 7) * 32;         // col base
    const long rowg = (long)blockIdx.x * MTILE + pr;

    const float dtv = dt[rowg] * (1.0f / NUM_UPDATES);

    // ---- prologue: stage x (bf16), y, z ----
    {
        const float* xr = x    + rowg * UD + pc0;
        const float* yr = y_in + rowg * UD + pc0;
        const float* zr = z_in + rowg * UD + pc0;
#pragma unroll 8
        for (int j = 0; j < 32; ++j) {
            float xv = xr[j], yv = yr[j], zv = zr[j];
            int o = pr * UD + pc0 + j;
            s_xbf[o] = (bf16_t)xv;
            s_y[o]   = yv;
            s_ybf[o] = (bf16_t)yv;
            s_z[o]   = zv;
        }
    }
    __syncthreads();

    // ---- GEMM1: s_i[32][1024] = xbf @ inp2hid (K=256), 8 N-tiles/wave ----
    gemm_tile<8>(s_xbf, wi, wave * 128, lane, [&](int n0, int m0, v8f acc) {
        store_c_bf16(s_i, m0, n0, 1024, lane, acc);
    });
    __syncthreads();

    // ---- 6 recurrent updates, fully in LDS ----
    for (int it = 0; it < NUM_UPDATES; ++it) {
        // GEMM2: s_th[32][768] = ybf @ hid2hid, 6 N-tiles/wave
        gemm_tile<6>(s_ybf, wh, wave * 96, lane, [&](int n0, int m0, v8f acc) {
            store_c_f32(s_th, m0, n0, 768, lane, acc);
        });
        __syncthreads();

        // pointwise z-update: z = (1-ms_dt)*z + ms_dt*tanh(i_y + h_y)
#pragma unroll 4
        for (int j = 0; j < 32; ++j) {
            const int c = pc0 + j;
            float i_dt2 = (float)s_i[pr * 1024 + 256 + c];
            float h_dt2 = s_th[pr * 768 + 256 + c];
            float i_y   = (float)s_i[pr * 1024 + 768 + c];
            float h_y   = s_th[pr * 768 + 512 + c];
            float ms    = dtv * sigmoidf_(i_dt2 + h_dt2);
            float zv    = s_z[pr * UD + c];
            float zn    = (1.0f - ms) * zv + ms * tanhf(i_y + h_y);
            s_z[pr * UD + c]   = zn;
            s_zbf[pr * UD + c] = (bf16_t)zn;
        }
        __syncthreads();

        // GEMM3: s_tz[32][256] = zbf @ transform_z, 2 N-tiles/wave
        gemm_tile<2>(s_zbf, wt, wave * 32, lane, [&](int n0, int m0, v8f acc) {
            store_c_f32(s_tz, m0, n0, UD, lane, acc);
        });
        __syncthreads();

        // pointwise y-update: y = (1-ms_bar)*y + ms_bar*tanh(z@Tz + i_z)
#pragma unroll 4
        for (int j = 0; j < 32; ++j) {
            const int c = pc0 + j;
            float i_dt1 = (float)s_i[pr * 1024 + c];
            float h_dt1 = s_th[pr * 768 + c];
            float i_z   = (float)s_i[pr * 1024 + 512 + c];
            float msb   = dtv * sigmoidf_(i_dt1 + h_dt1);
            float yv    = s_y[pr * UD + c];
            float yn = (1.0f - msb) * yv + msb * tanhf(s_tz[pr * UD + c] + i_z);
            s_y[pr * UD + c]   = yn;
            s_ybf[pr * UD + c] = (bf16_t)yn;
        }
        __syncthreads();
    }

    // ---- epilogue: write y, z ----
    {
        const long o = rowg * UD + pc0;
#pragma unroll 8
        for (int j = 0; j < 32; ++j) {
            out_y[o + j] = s_y[pr * UD + pc0 + j];
            out_z[o + j] = s_z[pr * UD + pc0 + j];
        }
    }
}

extern "C" void kernel_launch(void* const* d_in, const int* in_sizes, int n_in,
                              void* d_out, int out_size, void* d_ws, size_t ws_size,
                              hipStream_t stream) {
    const float* x   = (const float*)d_in[0];
    const float* dt  = (const float*)d_in[1];
    const float* y   = (const float*)d_in[2];
    const float* z   = (const float*)d_in[3];
    const float* i2h = (const float*)d_in[4];
    const float* h2h = (const float*)d_in[5];
    const float* tz  = (const float*)d_in[6];
    float* out = (float*)d_out;

    const long B = (long)in_sizes[0] / UD;   // 65536

    bf16_t* wi = (bf16_t*)d_ws;              // [1024][256] bf16
    bf16_t* wh = wi + 1024 * UD;             // [768][256]
    bf16_t* wt = wh + 768 * UD;              // [256][256]

    lem_prep_weights<<<2048, 256, 0, stream>>>(i2h, h2h, tz, wi, wh, wt);

    const size_t shmem = 288 * 1024;         // <= 320KB/WGP
    lem_main<<<dim3((unsigned)(B / MTILE)), dim3(256), shmem, stream>>>(
        x, dt, y, z, wi, wh, wt, out, out + B * UD);
}
